// RelationFeaturePropagation_46385646797139
// MI455X (gfx1250) — matrix operand order repaired and checked by reading, not compiled, not run on hardware
//
#include <hip/hip_runtime.h>
#include <hip/hip_bf16.h>
#include <math.h>

// Problem constants (match reference)
#define B_SZ   8
#define N_PTS  8192
#define S_PTS  2048
#define D_CH   256
#define M_ROWS (B_SZ * N_PTS)   // 65536 rows
#define K0_DIM (2 * D_CH)       // 512
#define C0_OUT 256
#define C1_OUT 128
#define EPS_BN 1e-5f

typedef __attribute__((ext_vector_type(16))) __bf16 v16bf;
typedef __attribute__((ext_vector_type(8)))  __bf16 v8bf;
typedef __attribute__((ext_vector_type(8)))  float  v8f;

union Frag16 { v16bf v; v8bf h[2]; };

__device__ __forceinline__ unsigned short f2bf_rne(float f) {
  union { float f; unsigned int u; } x; x.f = f;
  unsigned int r = x.u + 0x7FFFu + ((x.u >> 16) & 1u);
  return (unsigned short)(r >> 16);
}

// ---------------------------------------------------------------------------
// CDNA5 async global->LDS copy (ASYNCcnt-tracked, no VGPR round trip).
// Inline asm: VDST = LDS byte address (u32), VADDR = 64-bit global address.
// ---------------------------------------------------------------------------
__device__ __forceinline__ void async_copy_b128(const unsigned short* g,
                                                unsigned short* l) {
  asm volatile("global_load_async_to_lds_b128 %0, %1, off"
               :: "v"((unsigned)(uintptr_t)l), "v"(g)
               : "memory");
}

__device__ __forceinline__ void wait_async0() {
#if __has_builtin(__builtin_amdgcn_s_wait_asynccnt)
  __builtin_amdgcn_s_wait_asynccnt(0);
#else
  asm volatile("s_wait_asynccnt 0x0" ::: "memory");
#endif
}

// ---------------------------------------------------------------------------
// K0: fp32 -> bf16 weight conversion (W0, W1)
// ---------------------------------------------------------------------------
__global__ __launch_bounds__(256) void cvt_bf16_kernel(const float* __restrict__ src,
                                                       unsigned short* __restrict__ dst,
                                                       int count) {
  int i = blockIdx.x * 256 + threadIdx.x;
  if (i < count) dst[i] = f2bf_rne(src[i]);
}

// ---------------------------------------------------------------------------
// K1: 3-NN search + sigmoid relation weights (pre-divided by K=3)
// grid (N/256, B), block 256 (one thread per query point)
// ---------------------------------------------------------------------------
__global__ __launch_bounds__(256) void knn_weights_kernel(const float* __restrict__ xyz1,
                                                          const float* __restrict__ xyz2,
                                                          const float* __restrict__ rel_w,
                                                          const float* __restrict__ rel_b,
                                                          int*   __restrict__ idx_out,
                                                          float* __restrict__ w_out) {
  const int b   = blockIdx.y;
  const int n   = blockIdx.x * 256 + threadIdx.x;
  const int tid = threadIdx.x;

  const float px = xyz1[((size_t)b * N_PTS + n) * 3 + 0];
  const float py = xyz1[((size_t)b * N_PTS + n) * 3 + 1];
  const float pz = xyz1[((size_t)b * N_PTS + n) * 3 + 2];

  __shared__ float sx[256], sy[256], sz[256];

  float d0 = 3.4e38f, d1 = 3.4e38f, d2 = 3.4e38f;
  int   i0 = 0, i1 = 0, i2 = 0;
  float x0 = 0, y0 = 0, z0 = 0, x1 = 0, y1 = 0, z1 = 0, x2 = 0, y2 = 0, z2 = 0;

  for (int t = 0; t < S_PTS; t += 256) {
    const int s = t + tid;
    sx[tid] = xyz2[((size_t)b * S_PTS + s) * 3 + 0];
    sy[tid] = xyz2[((size_t)b * S_PTS + s) * 3 + 1];
    sz[tid] = xyz2[((size_t)b * S_PTS + s) * 3 + 2];
    __syncthreads();
    for (int j = 0; j < 256; ++j) {
      const float dx = px - sx[j];
      const float dy = py - sy[j];
      const float dz = pz - sz[j];
      const float d  = dx * dx + dy * dy + dz * dz;
      const int   si = t + j;
      if (d < d2) {
        if (d < d0) {
          d2 = d1; i2 = i1; x2 = x1; y2 = y1; z2 = z1;
          d1 = d0; i1 = i0; x1 = x0; y1 = y0; z1 = z0;
          d0 = d;  i0 = si; x0 = dx; y0 = dy; z0 = dz;
        } else if (d < d1) {
          d2 = d1; i2 = i1; x2 = x1; y2 = y1; z2 = z1;
          d1 = d;  i1 = si; x1 = dx; y1 = dy; z1 = dz;
        } else {
          d2 = d;  i2 = si; x2 = dx; y2 = dy; z2 = dz;
        }
      }
    }
    __syncthreads();
  }

  const float rw0 = rel_w[0], rw1 = rel_w[1], rw2 = rel_w[2], rw3 = rel_w[3];
  const float rb  = rel_b[0];
  const float w0f = (1.0f / 3.0f) / (1.0f + expf(-(d0 * rw0 + x0 * rw1 + y0 * rw2 + z0 * rw3 + rb)));
  const float w1f = (1.0f / 3.0f) / (1.0f + expf(-(d1 * rw0 + x1 * rw1 + y1 * rw2 + z1 * rw3 + rb)));
  const float w2f = (1.0f / 3.0f) / (1.0f + expf(-(d2 * rw0 + x2 * rw1 + y2 * rw2 + z2 * rw3 + rb)));

  const size_t base = ((size_t)b * N_PTS + n) * 3;
  idx_out[base + 0] = i0; idx_out[base + 1] = i1; idx_out[base + 2] = i2;
  w_out[base + 0] = w0f;  w_out[base + 1] = w1f;  w_out[base + 2] = w2f;
}

// ---------------------------------------------------------------------------
// K2: h[m, 0:256] = bf16(points1), h[m, 256:512] = bf16(weighted interp)
// grid (M_ROWS), block 256 (one thread per channel)
// ---------------------------------------------------------------------------
__global__ __launch_bounds__(256) void build_h_kernel(const float* __restrict__ points1,
                                                      const float* __restrict__ points2,
                                                      const int*   __restrict__ idx,
                                                      const float* __restrict__ w,
                                                      unsigned short* __restrict__ h) {
  const size_t m = blockIdx.x;
  const int    c = threadIdx.x;
  const int    b = (int)(m / N_PTS);

  h[m * K0_DIM + c] = f2bf_rne(points1[m * D_CH + c]);

  const size_t ib = m * 3;
  const float* p2 = points2 + (size_t)b * S_PTS * D_CH;
  const float  v  = w[ib + 0] * p2[(size_t)idx[ib + 0] * D_CH + c]
                  + w[ib + 1] * p2[(size_t)idx[ib + 1] * D_CH + c]
                  + w[ib + 2] * p2[(size_t)idx[ib + 2] * D_CH + c];
  h[m * K0_DIM + D_CH + c] = f2bf_rne(v);
}

// ---------------------------------------------------------------------------
// K3/K6: bf16 WMMA GEMM:  C[M, Ncols] = A[M, KDIM] * W[Ncols, KDIM]^T + bias
// block: 256 threads (8 waves), 128x128 tile; wave = 2x4 wmma tiles of 16x16
// Double-buffered LDS staged with GLOBAL_LOAD_ASYNC_TO_LDS_B128 (ASYNCcnt),
// padded stride (40 halves = 80B) for conflict-free fragment reads.
// ---------------------------------------------------------------------------
#define LDS_STRIDE 40

template <int KDIM>
__global__ __launch_bounds__(256) void gemm_bias_kernel(const unsigned short* __restrict__ A,
                                                        const unsigned short* __restrict__ W,
                                                        const float* __restrict__ bias,
                                                        float* __restrict__ C,
                                                        int Ncols) {
  __shared__ __align__(16) unsigned short lA[2][128 * LDS_STRIDE];
  __shared__ __align__(16) unsigned short lB[2][128 * LDS_STRIDE];

  const int tid   = threadIdx.x;
  const int lane  = tid & 31;
  const int wave  = tid >> 5;
  const int waveM = wave & 3;   // 4 waves along M
  const int waveN = wave >> 2;  // 2 waves along N
  const int lrow  = lane & 15;
  const int lhi   = lane >> 4;

  const int mBase = blockIdx.x * 128;
  const int nBase = blockIdx.y * 128;

  // stage one 128x32 bf16 tile of A and of W into LDS buffer `buf`
  auto stage = [&](int kt, int buf) {
    const int k0 = kt * 32;
#pragma unroll
    for (int i = 0; i < 2; ++i) {
      const int c  = tid + i * 256;  // 0..511 chunk id
      const int r  = c >> 2;         // row within tile
      const int ch = c & 3;          // 16B chunk within row
      async_copy_b128(A + (size_t)(mBase + r) * KDIM + k0 + ch * 8,
                      &lA[buf][r * LDS_STRIDE + ch * 8]);
      async_copy_b128(W + (size_t)(nBase + r) * KDIM + k0 + ch * 8,
                      &lB[buf][r * LDS_STRIDE + ch * 8]);
      if (k0 + 64 < KDIM) {  // prefetch tile kt+2 into near caches
        __builtin_prefetch(A + (size_t)(mBase + r) * KDIM + k0 + 64 + ch * 8, 0, 3);
        __builtin_prefetch(W + (size_t)(nBase + r) * KDIM + k0 + 64 + ch * 8, 0, 3);
      }
    }
  };

  v8f acc[2][4];
#pragma unroll
  for (int mt = 0; mt < 2; ++mt)
#pragma unroll
    for (int nt = 0; nt < 4; ++nt) acc[mt][nt] = (v8f)0.0f;

  constexpr int NK = KDIM / 32;
  stage(0, 0);
  wait_async0();
  __syncthreads();

  for (int kt = 0; kt < NK; ++kt) {
    const int buf = kt & 1;
    if (kt + 1 < NK) stage(kt + 1, buf ^ 1);

    // fragment loads per documented 16-bit A (16x32) / B (32x16) VGPR layouts
    Frag16 afrag[2], bfrag[4];
#pragma unroll
    for (int mt = 0; mt < 2; ++mt) {
      const int row = (waveM * 2 + mt) * 16 + lrow;
      afrag[mt].h[0] = *(const v8bf*)&lA[buf][row * LDS_STRIDE + lhi * 8];
      afrag[mt].h[1] = *(const v8bf*)&lA[buf][row * LDS_STRIDE + lhi * 8 + 16];
    }
#pragma unroll
    for (int nt = 0; nt < 4; ++nt) {
      const int row = (waveN * 4 + nt) * 16 + lrow;
      bfrag[nt].h[0] = *(const v8bf*)&lB[buf][row * LDS_STRIDE + lhi * 8];
      bfrag[nt].h[1] = *(const v8bf*)&lB[buf][row * LDS_STRIDE + lhi * 8 + 16];
    }

#pragma unroll
    for (int mt = 0; mt < 2; ++mt)
#pragma unroll
      for (int nt = 0; nt < 4; ++nt)
        acc[mt][nt] = __builtin_amdgcn_wmma_f32_16x16x32_bf16(
            false, afrag[mt].v, false, bfrag[nt].v, (short)0, acc[mt][nt],
            false, false);

    // all async writes (next tile) done + all waves past their reads
    wait_async0();
    __syncthreads();
  }

  // epilogue: +bias, fp32 store (C layout: VGPR r -> M = lhi*8 + r, N = lrow)
#pragma unroll
  for (int nt = 0; nt < 4; ++nt) {
    const int col = nBase + (waveN * 4 + nt) * 16 + lrow;
    const float bv = bias[col];
#pragma unroll
    for (int mt = 0; mt < 2; ++mt) {
#pragma unroll
      for (int r = 0; r < 8; ++r) {
        const int row = mBase + (waveM * 2 + mt) * 16 + lhi * 8 + r;
        C[(size_t)row * Ncols + col] = acc[mt][nt][r] + bv;
      }
    }
  }
}

// ---------------------------------------------------------------------------
// K4/K7: per-channel training-mode BN statistics -> (scale, shift)
// grid (Ncols), block 256
// ---------------------------------------------------------------------------
__global__ __launch_bounds__(256) void col_stats_kernel(const float* __restrict__ Y,
                                                        int Ncols,
                                                        const float* __restrict__ g,
                                                        const float* __restrict__ be,
                                                        float* __restrict__ scale,
                                                        float* __restrict__ shift) {
  const int c = blockIdx.x;
  float s = 0.f, s2 = 0.f;
  for (int r = threadIdx.x; r < M_ROWS; r += 256) {
    const float v = Y[(size_t)r * Ncols + c];
    s += v; s2 += v * v;
  }
  __shared__ float sh[256], sh2[256];
  sh[threadIdx.x] = s; sh2[threadIdx.x] = s2;
  __syncthreads();
  for (int off = 128; off > 0; off >>= 1) {
    if (threadIdx.x < off) {
      sh[threadIdx.x]  += sh[threadIdx.x + off];
      sh2[threadIdx.x] += sh2[threadIdx.x + off];
    }
    __syncthreads();
  }
  if (threadIdx.x == 0) {
    const float inv_m = 1.0f / (float)M_ROWS;
    const float mean  = sh[0] * inv_m;
    const float var   = sh2[0] * inv_m - mean * mean;
    const float sc    = g[c] * rsqrtf(var + EPS_BN);
    scale[c] = sc;
    shift[c] = be[c] - mean * sc;
  }
}

// ---------------------------------------------------------------------------
// K5: BN + ReLU -> bf16 (feeds next GEMM).   grid (M_ROWS), block Ncols
// ---------------------------------------------------------------------------
__global__ void bn_relu_bf16_kernel(const float* __restrict__ Y,
                                    const float* __restrict__ scale,
                                    const float* __restrict__ shift,
                                    unsigned short* __restrict__ X,
                                    int Ncols) {
  const size_t m = blockIdx.x;
  const int    c = threadIdx.x;
  const float  v = fmaxf(fmaf(Y[m * Ncols + c], scale[c], shift[c]), 0.0f);
  X[m * Ncols + c] = f2bf_rne(v);
}

// ---------------------------------------------------------------------------
// K8: BN + ReLU -> fp32 final output.   grid (M_ROWS), block Ncols
// ---------------------------------------------------------------------------
__global__ void bn_relu_f32_kernel(const float* __restrict__ Y,
                                   const float* __restrict__ scale,
                                   const float* __restrict__ shift,
                                   float* __restrict__ O,
                                   int Ncols) {
  const size_t m = blockIdx.x;
  const int    c = threadIdx.x;
  O[m * Ncols + c] = fmaxf(fmaf(Y[m * Ncols + c], scale[c], shift[c]), 0.0f);
}

// ---------------------------------------------------------------------------
extern "C" void kernel_launch(void* const* d_in, const int* in_sizes, int n_in,
                              void* d_out, int out_size, void* d_ws, size_t ws_size,
                              hipStream_t stream) {
  const float* xyz1    = (const float*)d_in[0];
  const float* xyz2    = (const float*)d_in[1];
  const float* points1 = (const float*)d_in[2];
  const float* points2 = (const float*)d_in[3];
  const float* rel_w   = (const float*)d_in[4];
  const float* rel_b   = (const float*)d_in[5];
  const float* W0      = (const float*)d_in[6];
  const float* b0      = (const float*)d_in[7];
  const float* g0      = (const float*)d_in[8];
  const float* be0     = (const float*)d_in[9];
  const float* W1      = (const float*)d_in[10];
  const float* b1      = (const float*)d_in[11];
  const float* g1      = (const float*)d_in[12];
  const float* be1     = (const float*)d_in[13];
  float* out = (float*)d_out;

  // workspace layout (256B aligned slabs)
  char*  ws  = (char*)d_ws;
  size_t off = 0;
  auto slab = [&](size_t bytes) -> char* {
    char* p = ws + off;
    off = (off + bytes + 255) & ~(size_t)255;
    return p;
  };
  unsigned short* w0b = (unsigned short*)slab((size_t)C0_OUT * K0_DIM * 2);
  unsigned short* w1b = (unsigned short*)slab((size_t)C1_OUT * C0_OUT * 2);
  int*            idx = (int*)           slab((size_t)M_ROWS * 3 * 4);
  float*          wgt = (float*)         slab((size_t)M_ROWS * 3 * 4);
  unsigned short* h   = (unsigned short*)slab((size_t)M_ROWS * K0_DIM * 2);
  float*          y0  = (float*)         slab((size_t)M_ROWS * C0_OUT * 4);
  unsigned short* x1  = (unsigned short*)slab((size_t)M_ROWS * C0_OUT * 2);
  float*          y1  = (float*)         slab((size_t)M_ROWS * C1_OUT * 4);
  float*          sc0 = (float*)slab(C0_OUT * 4);
  float*          sh0 = (float*)slab(C0_OUT * 4);
  float*          sc1 = (float*)slab(C1_OUT * 4);
  float*          sh1 = (float*)slab(C1_OUT * 4);

  // weights -> bf16
  cvt_bf16_kernel<<<(C0_OUT * K0_DIM + 255) / 256, 256, 0, stream>>>(W0, w0b, C0_OUT * K0_DIM);
  cvt_bf16_kernel<<<(C1_OUT * C0_OUT + 255) / 256, 256, 0, stream>>>(W1, w1b, C1_OUT * C0_OUT);

  // 3-NN + relation weights
  knn_weights_kernel<<<dim3(N_PTS / 256, B_SZ), 256, 0, stream>>>(
      xyz1, xyz2, rel_w, rel_b, idx, wgt);

  // h = concat(points1, interp) in bf16
  build_h_kernel<<<M_ROWS, 256, 0, stream>>>(points1, points2, idx, wgt, h);

  // layer 0: GEMM + bias, stats, BN+ReLU -> bf16
  gemm_bias_kernel<K0_DIM><<<dim3(M_ROWS / 128, C0_OUT / 128), 256, 0, stream>>>(
      h, w0b, b0, y0, C0_OUT);
  col_stats_kernel<<<C0_OUT, 256, 0, stream>>>(y0, C0_OUT, g0, be0, sc0, sh0);
  bn_relu_bf16_kernel<<<M_ROWS, C0_OUT, 0, stream>>>(y0, sc0, sh0, x1, C0_OUT);

  // layer 1: GEMM + bias, stats, BN+ReLU -> fp32 out
  gemm_bias_kernel<C0_OUT><<<dim3(M_ROWS / 128, C1_OUT / 128), 256, 0, stream>>>(
      x1, w1b, b1, y1, C1_OUT);
  col_stats_kernel<<<C1_OUT, 256, 0, stream>>>(y1, C1_OUT, g1, be1, sc1, sh1);
  bn_relu_f32_kernel<<<M_ROWS, C1_OUT, 0, stream>>>(y1, sc1, sh1, out, C1_OUT);
}
